// CrossAttention_63866163691863
// MI455X (gfx1250) — matrix operand order repaired and checked by reading, not compiled, not run on hardware
//
#include <hip/hip_runtime.h>

#define DM    768
#define NSEQ  1024
#define BATCH 8
#define NHEAD 12
#define DH    64
#define NTOK  (BATCH * NSEQ)   // 8192
#define LNEPS 1e-5f

typedef __bf16 bf16_t;
typedef __bf16 v16bf __attribute__((ext_vector_type(16)));
typedef float  v8f   __attribute__((ext_vector_type(8)));

union Frag16 { v16bf v; unsigned u[8]; };

// A-matrix (16x32, 16-bit) K offset for packed pair i (0..7), lane-half h.
__device__ __forceinline__ int a_koff(int i, int h) {
  return ((i >> 2) << 4) + (h << 3) + ((i & 3) << 1);
}
// B-matrix (32x16, 16-bit) K offset for packed pair i (0..7), lane-half h.
__device__ __forceinline__ int b_koff(int i, int h) {
  return ((i >> 2) << 4) + (h << 3) + ((i & 3) << 1);
}

__device__ __forceinline__ v8f wmma_bf16(const Frag16& a, const Frag16& b, v8f c) {
  return __builtin_amdgcn_wmma_f32_16x16x32_bf16(false, a.v, false, b.v,
                                                 (short)0, c, false, false);
}

// CDNA5 async global->LDS copy (ASYNCcnt-tracked, no VGPR round trip).
__device__ __forceinline__ void async_load_b128(unsigned lds_off, const bf16_t* g) {
  asm volatile("global_load_async_to_lds_b128 %0, %1, off"
               :: "v"(lds_off), "v"(g) : "memory");
}
__device__ __forceinline__ void wait_async0() {
  asm volatile("s_wait_asynccnt 0" ::: "memory");
}
__device__ __forceinline__ unsigned lds_addr32(const void* p) {
  return (unsigned)(uintptr_t)p;   // LDS aperture: addr[31:0] is the LDS offset
}

// ---------------------------------------------------------------------------
// Kernel 1: convert + transpose weights: Wt[out][in] = bf16(W[in][out])
// ---------------------------------------------------------------------------
struct WPrepArgs { const float* w[8]; bf16_t* wt[8]; };

__global__ __launch_bounds__(256) void wprep_kernel(WPrepArgs a) {
  const float* w = a.w[blockIdx.z];
  bf16_t* wt = a.wt[blockIdx.z];
  __shared__ float tile[32][33];
  int i0 = blockIdx.y * 32, o0 = blockIdx.x * 32;
  #pragma unroll
  for (int r = 0; r < 4; ++r) {
    int i = threadIdx.y + r * 8;
    tile[i][threadIdx.x] = w[(size_t)(i0 + i) * DM + o0 + threadIdx.x];
  }
  __syncthreads();
  #pragma unroll
  for (int r = 0; r < 4; ++r) {
    int o = threadIdx.y + r * 8;
    wt[(size_t)(o0 + o) * DM + i0 + threadIdx.x] = (bf16_t)tile[threadIdx.x][o];
  }
}

// ---------------------------------------------------------------------------
// Kernel 2: LayerNorm row (768) -> bf16
// ---------------------------------------------------------------------------
__global__ __launch_bounds__(256) void layernorm_kernel(const float* __restrict__ x,
                                                        const float* __restrict__ g,
                                                        const float* __restrict__ be,
                                                        bf16_t* __restrict__ out) {
  int row = blockIdx.x;
  const float* xr = x + (size_t)row * DM;
  int tid = threadIdx.x, lane = tid & 31, wave = tid >> 5;
  float s = 0.f, s2 = 0.f;
  for (int i = tid; i < DM; i += 256) { float v = xr[i]; s += v; s2 += v * v; }
  #pragma unroll
  for (int off = 16; off > 0; off >>= 1) {
    s  += __shfl_xor(s, off, 32);
    s2 += __shfl_xor(s2, off, 32);
  }
  __shared__ float rs[8], rs2[8];
  if (lane == 0) { rs[wave] = s; rs2[wave] = s2; }
  __syncthreads();
  __shared__ float mu_s, rstd_s;
  if (tid == 0) {
    float a = 0.f, b = 0.f;
    #pragma unroll
    for (int i = 0; i < 8; ++i) { a += rs[i]; b += rs2[i]; }
    float mu = a / (float)DM;
    float var = b / (float)DM - mu * mu;
    mu_s = mu; rstd_s = rsqrtf(var + LNEPS);
  }
  __syncthreads();
  float mu = mu_s, rstd = rstd_s;
  bf16_t* o = out + (size_t)row * DM;
  for (int i = tid; i < DM; i += 256)
    o[i] = (bf16_t)((xr[i] - mu) * rstd * g[i] + be[i]);
}

// ---------------------------------------------------------------------------
// Kernel 3: QKV projections. C[8192,768] = m[8192,768] x Wt^T + bias.
// Double-buffered LDS filled via async global->LDS copies.
// Q/K stored [B,H,N,Dh]; V stored transposed [B,H,Dh,N].
// ---------------------------------------------------------------------------
struct ProjArgs {
  const bf16_t* src[6];
  const bf16_t* wt[6];
  const float*  bias[6];
  bf16_t*       dst[6];
  int           vflag[6];
};

__global__ __launch_bounds__(256) void qkv_proj_kernel(ProjArgs args) {
  const int mode = blockIdx.z;
  const bf16_t* __restrict__ src = args.src[mode];
  const bf16_t* __restrict__ wt  = args.wt[mode];
  const float*  __restrict__ bias = args.bias[mode];
  bf16_t* __restrict__ dst = args.dst[mode];
  const int vstore = args.vflag[mode];

  __shared__ __attribute__((aligned(16))) bf16_t As[2][128][32];  // 16 KB
  __shared__ __attribute__((aligned(16))) bf16_t Bs[2][64][32];   //  8 KB

  const int tid = threadIdx.x, lane = tid & 31, wave = tid >> 5;
  const int wm = wave >> 1, wn = wave & 1;
  const int half = lane >> 4, col = lane & 15;
  const int block_m = blockIdx.x * 128, block_n = blockIdx.y * 64;

  const unsigned asBase = lds_addr32(&As[0][0][0]);
  const unsigned bsBase = lds_addr32(&Bs[0][0][0]);
  const int arow = tid >> 2, aseg = tid & 3;         // A: 2 x (row,seg) per thread
  const int arow2 = (tid + 256) >> 2, aseg2 = (tid + 256) & 3;
  const int brow = tid >> 2, bseg = tid & 3;         // B: 1 per thread

  auto issue_tile = [&](int buf, int k0) {
    async_load_b128(asBase + (unsigned)(buf * 8192 + arow * 64 + aseg * 16),
                    src + (size_t)(block_m + arow) * DM + k0 + aseg * 8);
    async_load_b128(asBase + (unsigned)(buf * 8192 + arow2 * 64 + aseg2 * 16),
                    src + (size_t)(block_m + arow2) * DM + k0 + aseg2 * 8);
    async_load_b128(bsBase + (unsigned)(buf * 4096 + brow * 64 + bseg * 16),
                    wt + (size_t)(block_n + brow) * DM + k0 + bseg * 8);
  };

  v8f acc[2][2] = {};
  issue_tile(0, 0);
  for (int k0 = 0; k0 < DM; k0 += 32) {
    const int buf = (k0 >> 5) & 1;
    wait_async0();
    __syncthreads();
    if (k0 + 32 < DM) issue_tile(buf ^ 1, k0 + 32);

    Frag16 a[2], b[2];
    #pragma unroll
    for (int s = 0; s < 2; ++s)
      #pragma unroll
      for (int i = 0; i < 8; ++i)
        a[s].u[i] = *(const unsigned*)(&As[buf][wm * 32 + s * 16 + col][a_koff(i, half)]);
    #pragma unroll
    for (int t = 0; t < 2; ++t)
      #pragma unroll
      for (int i = 0; i < 8; ++i)
        b[t].u[i] = *(const unsigned*)(&Bs[buf][wn * 32 + t * 16 + col][b_koff(i, half)]);

    #pragma unroll
    for (int s = 0; s < 2; ++s)
      #pragma unroll
      for (int t = 0; t < 2; ++t)
        acc[s][t] = wmma_bf16(a[s], b[t], acc[s][t]);
  }

  #pragma unroll
  for (int s = 0; s < 2; ++s)
    #pragma unroll
    for (int t = 0; t < 2; ++t) {
      int o = block_n + wn * 32 + t * 16 + col;
      float bo = bias[o];
      int h = o >> 6, d = o & 63;
      #pragma unroll
      for (int r = 0; r < 8; ++r) {
        int gm = block_m + wm * 32 + s * 16 + r + 8 * half;
        int bi = gm >> 10, n = gm & 1023;
        float val = acc[s][t][r] + bo;
        if (vstore)
          dst[(((size_t)bi * NHEAD + h) * DH + d) * NSEQ + n] = (bf16_t)val;
        else
          dst[(((size_t)bi * NHEAD + h) * NSEQ + n) * DH + d] = (bf16_t)val;
      }
    }
}

// ---------------------------------------------------------------------------
// Kernel 4: flash-style cross-attention. One wave = 16 query rows.
// ---------------------------------------------------------------------------
struct AttnArgs {
  const bf16_t* q[2];
  const bf16_t* k[2];
  const bf16_t* vt[2];
  bf16_t*       out[2];
};

__global__ __launch_bounds__(256) void attn_kernel(AttnArgs args) {
  const int pair = blockIdx.z;
  const int bh = blockIdx.y;
  const int b = bh / NHEAD, h = bh % NHEAD;
  const bf16_t* __restrict__ Q  = args.q[pair]  + (size_t)bh * NSEQ * DH;
  const bf16_t* __restrict__ K  = args.k[pair]  + (size_t)bh * NSEQ * DH;
  const bf16_t* __restrict__ Vt = args.vt[pair] + (size_t)bh * DH * NSEQ;
  bf16_t* __restrict__ Out = args.out[pair];

  __shared__ bf16_t Pld[8][16][32];   // per-wave P staging, 8 KB total

  const int lane = threadIdx.x & 31, wave = threadIdx.x >> 5;
  const int half = lane >> 4, col = lane & 15;
  const int qbase = blockIdx.x * 128 + wave * 16;

  // Q fragments (16 rows x 64 dh = two A frags), kept resident
  Frag16 aq[2];
  #pragma unroll
  for (int kk = 0; kk < 2; ++kk)
    #pragma unroll
    for (int i = 0; i < 8; ++i)
      aq[kk].u[i] = *(const unsigned*)(Q + (size_t)(qbase + col) * DH +
                                       kk * 32 + a_koff(i, half));

  float m_i[8], l_i[8];
  #pragma unroll
  for (int r = 0; r < 8; ++r) { m_i[r] = -1e30f; l_i[r] = 0.f; }
  v8f o_acc[4] = {};

  for (int kc = 0; kc < NSEQ / 32; ++kc) {
    const int key0 = kc * 32;
    // ----- S = Q K^T for 32 keys (two 16-col tiles, chained over Dh) -----
    v8f sfrag[2];
    #pragma unroll
    for (int t = 0; t < 2; ++t) {
      v8f c = 0;
      #pragma unroll
      for (int kk = 0; kk < 2; ++kk) {
        Frag16 bk;
        #pragma unroll
        for (int i = 0; i < 8; ++i) {
          int key = key0 + t * 16 + col;
          bk.u[i] = *(const unsigned*)(K + (size_t)key * DH + kk * 32 + b_koff(i, half));
        }
        c = wmma_bf16(aq[kk], bk, c);
      }
      sfrag[t] = c;
    }
    // ----- online softmax over this chunk -----
    float alpha[8], p0[8], p1[8];
    #pragma unroll
    for (int r = 0; r < 8; ++r) {
      float s0 = sfrag[0][r] * 0.125f;        // 1/sqrt(64)
      float s1 = sfrag[1][r] * 0.125f;
      float mx = fmaxf(s0, s1);
      #pragma unroll
      for (int off = 1; off < 16; off <<= 1)
        mx = fmaxf(mx, __shfl_xor(mx, off, 32));
      float mnew = fmaxf(m_i[r], mx);
      alpha[r] = __expf(m_i[r] - mnew);
      p0[r] = __expf(s0 - mnew);
      p1[r] = __expf(s1 - mnew);
      float ps = p0[r] + p1[r];
      #pragma unroll
      for (int off = 1; off < 16; off <<= 1)
        ps += __shfl_xor(ps, off, 32);
      l_i[r] = l_i[r] * alpha[r] + ps;
      m_i[r] = mnew;
    }
    #pragma unroll
    for (int t = 0; t < 4; ++t)
      #pragma unroll
      for (int r = 0; r < 8; ++r)
        o_acc[t][r] *= alpha[r];
    // ----- stage P (C layout -> row-major LDS -> A layout) -----
    #pragma unroll
    for (int r = 0; r < 8; ++r) {
      Pld[wave][r + 8 * half][col]      = (bf16_t)p0[r];
      Pld[wave][r + 8 * half][col + 16] = (bf16_t)p1[r];
    }
    asm volatile("s_wait_dscnt 0" ::: "memory");
    Frag16 pa;
    #pragma unroll
    for (int i = 0; i < 8; ++i)
      pa.u[i] = *(const unsigned*)(&Pld[wave][col][a_koff(i, half)]);
    // ----- O += P V (4 column tiles over Dh=64) -----
    #pragma unroll
    for (int t = 0; t < 4; ++t) {
      Frag16 bv;
      #pragma unroll
      for (int i = 0; i < 8; ++i) {
        int d = t * 16 + col;
        bv.u[i] = *(const unsigned*)(Vt + (size_t)d * NSEQ + key0 + b_koff(i, half));
      }
      o_acc[t] = wmma_bf16(pa, bv, o_acc[t]);
    }
  }

  float inv_l[8];
  #pragma unroll
  for (int r = 0; r < 8; ++r) inv_l[r] = 1.f / l_i[r];
  #pragma unroll
  for (int t = 0; t < 4; ++t)
    #pragma unroll
    for (int r = 0; r < 8; ++r) {
      int n = qbase + r + 8 * half;
      int d = t * 16 + col;
      Out[((size_t)b * NSEQ + n) * DM + h * DH + d] = (bf16_t)(o_acc[t][r] * inv_l[r]);
    }
}

// ---------------------------------------------------------------------------
// Kernel 5: output projection -> f32 d_out (async double-buffered like kernel 3)
// ---------------------------------------------------------------------------
struct OutArgs {
  const bf16_t* src[2];
  const bf16_t* wt[2];
  const float*  bias[2];
  float*        dst[2];
};

__global__ __launch_bounds__(256) void out_proj_kernel(OutArgs args) {
  const int mode = blockIdx.z;
  const bf16_t* __restrict__ src = args.src[mode];
  const bf16_t* __restrict__ wt  = args.wt[mode];
  const float*  __restrict__ bias = args.bias[mode];
  float* __restrict__ dst = args.dst[mode];

  __shared__ __attribute__((aligned(16))) bf16_t As[2][128][32];
  __shared__ __attribute__((aligned(16))) bf16_t Bs[2][64][32];

  const int tid = threadIdx.x, lane = tid & 31, wave = tid >> 5;
  const int wm = wave >> 1, wn = wave & 1;
  const int half = lane >> 4, col = lane & 15;
  const int block_m = blockIdx.x * 128, block_n = blockIdx.y * 64;

  const unsigned asBase = lds_addr32(&As[0][0][0]);
  const unsigned bsBase = lds_addr32(&Bs[0][0][0]);
  const int arow = tid >> 2, aseg = tid & 3;
  const int arow2 = (tid + 256) >> 2, aseg2 = (tid + 256) & 3;
  const int brow = tid >> 2, bseg = tid & 3;

  auto issue_tile = [&](int buf, int k0) {
    async_load_b128(asBase + (unsigned)(buf * 8192 + arow * 64 + aseg * 16),
                    src + (size_t)(block_m + arow) * DM + k0 + aseg * 8);
    async_load_b128(asBase + (unsigned)(buf * 8192 + arow2 * 64 + aseg2 * 16),
                    src + (size_t)(block_m + arow2) * DM + k0 + aseg2 * 8);
    async_load_b128(bsBase + (unsigned)(buf * 4096 + brow * 64 + bseg * 16),
                    wt + (size_t)(block_n + brow) * DM + k0 + bseg * 8);
  };

  v8f acc[2][2] = {};
  issue_tile(0, 0);
  for (int k0 = 0; k0 < DM; k0 += 32) {
    const int buf = (k0 >> 5) & 1;
    wait_async0();
    __syncthreads();
    if (k0 + 32 < DM) issue_tile(buf ^ 1, k0 + 32);

    Frag16 a[2], b[2];
    #pragma unroll
    for (int s = 0; s < 2; ++s)
      #pragma unroll
      for (int i = 0; i < 8; ++i)
        a[s].u[i] = *(const unsigned*)(&As[buf][wm * 32 + s * 16 + col][a_koff(i, half)]);
    #pragma unroll
    for (int t = 0; t < 2; ++t)
      #pragma unroll
      for (int i = 0; i < 8; ++i)
        b[t].u[i] = *(const unsigned*)(&Bs[buf][wn * 32 + t * 16 + col][b_koff(i, half)]);

    #pragma unroll
    for (int s = 0; s < 2; ++s)
      #pragma unroll
      for (int t = 0; t < 2; ++t)
        acc[s][t] = wmma_bf16(a[s], b[t], acc[s][t]);
  }

  #pragma unroll
  for (int s = 0; s < 2; ++s)
    #pragma unroll
    for (int t = 0; t < 2; ++t) {
      int o = block_n + wn * 32 + t * 16 + col;
      float bo = bias[o];
      #pragma unroll
      for (int r = 0; r < 8; ++r) {
        int gm = block_m + wm * 32 + s * 16 + r + 8 * half;
        dst[(size_t)gm * DM + o] = acc[s][t][r] + bo;
      }
    }
}

// ---------------------------------------------------------------------------
// Launch
// ---------------------------------------------------------------------------
extern "C" void kernel_launch(void* const* d_in, const int* in_sizes, int n_in,
                              void* d_out, int out_size, void* d_ws, size_t ws_size,
                              hipStream_t stream) {
  const float* x1   = (const float*)d_in[0];
  const float* x2   = (const float*)d_in[1];
  const float* ln1g = (const float*)d_in[2];
  const float* ln1b = (const float*)d_in[3];
  const float* ln2g = (const float*)d_in[4];
  const float* ln2b = (const float*)d_in[5];
  // weights at 6,8,10,12,14,16,18,20 ; biases at 7,9,11,13,15,17,19,21

  char* ws = (char*)d_ws;
  size_t off = 0;
  auto alloc_bf = [&](size_t elems) -> bf16_t* {
    bf16_t* p = (bf16_t*)(ws + off);
    off += elems * sizeof(bf16_t);
    return p;
  };
  bf16_t* wt[8];
  for (int i = 0; i < 8; ++i) wt[i] = alloc_bf((size_t)DM * DM);
  bf16_t* m1 = alloc_bf((size_t)NTOK * DM);
  bf16_t* m2 = alloc_bf((size_t)NTOK * DM);
  const size_t HSZ = (size_t)BATCH * NHEAD * NSEQ * DH;
  bf16_t* q1  = alloc_bf(HSZ);
  bf16_t* k1  = alloc_bf(HSZ);
  bf16_t* v1t = alloc_bf(HSZ);
  bf16_t* q2  = alloc_bf(HSZ);
  bf16_t* k2  = alloc_bf(HSZ);
  bf16_t* v2t = alloc_bf(HSZ);
  bf16_t* attn1 = alloc_bf((size_t)NTOK * DM);
  bf16_t* attn2 = alloc_bf((size_t)NTOK * DM);

  WPrepArgs wa;
  for (int i = 0; i < 8; ++i) { wa.w[i] = (const float*)d_in[6 + 2 * i]; wa.wt[i] = wt[i]; }
  wprep_kernel<<<dim3(24, 24, 8), dim3(32, 8), 0, stream>>>(wa);

  layernorm_kernel<<<NTOK, 256, 0, stream>>>(x1, ln1g, ln1b, m1);
  layernorm_kernel<<<NTOK, 256, 0, stream>>>(x2, ln2g, ln2b, m2);

  ProjArgs pa;
  const bf16_t* srcs[6] = { m1, m1, m1, m2, m2, m2 };
  int widx[6] = { 0, 1, 2, 3, 4, 5 };            // wq1,wk1,wv1,wq2,wk2,wv2
  int bidx[6] = { 7, 9, 11, 13, 15, 17 };
  bf16_t* dsts[6] = { q1, k1, v1t, q2, k2, v2t };
  int vfl[6] = { 0, 0, 1, 0, 0, 1 };
  for (int i = 0; i < 6; ++i) {
    pa.src[i] = srcs[i]; pa.wt[i] = wt[widx[i]];
    pa.bias[i] = (const float*)d_in[bidx[i]];
    pa.dst[i] = dsts[i]; pa.vflag[i] = vfl[i];
  }
  qkv_proj_kernel<<<dim3(64, 12, 6), 256, 0, stream>>>(pa);

  AttnArgs aa;
  aa.q[0] = q1;  aa.k[0] = k2; aa.vt[0] = v2t; aa.out[0] = attn1;  // out1 = attend(q1,k2,v2)
  aa.q[1] = q2;  aa.k[1] = k1; aa.vt[1] = v1t; aa.out[1] = attn2;  // out2 = attend(q2,k1,v1)
  attn_kernel<<<dim3(8, 96, 2), 256, 0, stream>>>(aa);

  OutArgs oa;
  oa.src[0] = attn1; oa.wt[0] = wt[6]; oa.bias[0] = (const float*)d_in[19];
  oa.dst[0] = (float*)d_out;
  oa.src[1] = attn2; oa.wt[1] = wt[7]; oa.bias[1] = (const float*)d_in[21];
  oa.dst[1] = (float*)d_out + (size_t)NTOK * DM;
  out_proj_kernel<<<dim3(64, 12, 2), 256, 0, stream>>>(oa);
}
